// SlotAttention_42640435315508
// MI455X (gfx1250) — matrix operand order repaired and checked by reading, not compiled, not run on hardware
//
#include <hip/hip_runtime.h>
#include <stdint.h>

// ---------------------------------------------------------------------------
// Problem constants (SlotAttention): B=64, N=4096, D=256, NS=8, ITERS=3
// ---------------------------------------------------------------------------
#define BB_ 64
#define NN_ 4096
#define DD_ 256
#define NSL 8

typedef __bf16 v16bf __attribute__((ext_vector_type(16)));
typedef float  v8f   __attribute__((ext_vector_type(8)));

union FragBF {
    v16bf v;
    uint4 q[2];
    unsigned short h[16];
};

__device__ inline v8f vzero() {
    v8f z = {0.f,0.f,0.f,0.f,0.f,0.f,0.f,0.f};
    return z;
}

__device__ inline unsigned short f2bf(float f) {
    unsigned int u = __float_as_uint(f);
    u += 0x7fffu + ((u >> 16) & 1u);          // round-to-nearest-even
    return (unsigned short)(u >> 16);
}

__device__ inline float wave_sum(float v) {
    #pragma unroll
    for (int m = 16; m >= 1; m >>= 1) v += __shfl_xor(v, m, 32);
    return v;
}

// A fragment (16x32 bf16), source row-major [row][K], row = lane&15.
// Lanes 0-15: K {0..7, 16..23};  lanes 16-31: K {8..15, 24..31}.
__device__ inline v16bf load_fragA(const unsigned short* __restrict__ base,
                                   int ld, int kbase) {
    int lane = threadIdx.x & 31;
    const unsigned short* p = base + (size_t)(lane & 15) * ld + kbase +
                              ((lane < 16) ? 0 : 8);
    FragBF f;
    f.q[0] = *(const uint4*)p;
    f.q[1] = *(const uint4*)(p + 16);
    return f.v;
}

// B fragment (32x16 bf16), source stored as Bt row-major [n][K] (i.e. W[n][k]),
// n = lane&15.  Lanes 0-15: K {0..7, 16..23};  lanes 16-31: K {8..15, 24..31}.
__device__ inline v16bf load_fragB(const unsigned short* __restrict__ base,
                                   int ld, int kbase) {
    int lane = threadIdx.x & 31;
    const unsigned short* p = base + (size_t)(lane & 15) * ld + kbase +
                              ((lane < 16) ? 0 : 8);
    FragBF f;
    f.q[0] = *(const uint4*)p;
    f.q[1] = *(const uint4*)(p + 16);
    return f.v;
}

__device__ inline v8f wmma_bf16(v16bf a, v16bf b, v8f c) {
    return __builtin_amdgcn_wmma_f32_16x16x32_bf16(
        false, a, false, b, (short)0, c, false, false);
}

// ---------------------------------------------------------------------------
// Setup kernels
// ---------------------------------------------------------------------------
__global__ void cvt_bf16_kernel(const float* __restrict__ src,
                                unsigned short* __restrict__ dst, int n) {
    int i = blockIdx.x * 256 + threadIdx.x;
    if (i < n) dst[i] = f2bf(src[i]);
}

__global__ void bcast_slots_kernel(const float* __restrict__ si,
                                   float* __restrict__ slots) {
    int i = blockIdx.x * 256 + threadIdx.x;       // 0 .. 131071
    slots[i] = si[i & (NSL * DD_ - 1)];
}

// ---------------------------------------------------------------------------
// Kernel 1: x_ln = LayerNorm(inputs); K = x_ln @ Wk^T; V = x_ln @ Wv^T (bf16)
// Grid: (B*N)/64 blocks, 128 threads (4 waves). 64-row M-slab per block.
// ---------------------------------------------------------------------------
__global__ __launch_bounds__(128) void ln_kv_kernel(
    const float* __restrict__ x, const float* __restrict__ g,
    const float* __restrict__ be,
    const unsigned short* __restrict__ wk, const unsigned short* __restrict__ wv,
    unsigned short* __restrict__ kout, unsigned short* __restrict__ vout)
{
    __shared__ unsigned short xt[64 * 264];       // 64 rows, padded stride
    const int wave = threadIdx.x >> 5, lane = threadIdx.x & 31;
    const size_t rowbase = (size_t)blockIdx.x * 64;

    // -------- LayerNorm 16 rows per wave, result bf16 into LDS ------------
    for (int t = 0; t < 16; ++t) {
        const int lr = wave * 16 + t;
        const float* xr = x + (rowbase + lr) * DD_;
        float4 a = ((const float4*)xr)[lane * 2];
        float4 c = ((const float4*)xr)[lane * 2 + 1];
        float vals[8] = {a.x, a.y, a.z, a.w, c.x, c.y, c.z, c.w};
        float s = 0.f, s2 = 0.f;
        #pragma unroll
        for (int e = 0; e < 8; ++e) { s += vals[e]; s2 += vals[e] * vals[e]; }
        s = wave_sum(s); s2 = wave_sum(s2);
        const float mean = s * (1.f / 256.f);
        const float inv  = rsqrtf(s2 * (1.f / 256.f) - mean * mean + 1e-5f);
        unsigned short o[8];
        #pragma unroll
        for (int e = 0; e < 8; ++e)
            o[e] = f2bf((vals[e] - mean) * inv * g[lane * 8 + e] + be[lane * 8 + e]);
        *(uint4*)&xt[lr * 264 + lane * 8] = *(const uint4*)o;
    }
    __syncthreads();

    // -------- WMMA: 16-row slab per wave x 32 n-tiles (16 K cols, 16 V) ---
    for (int nt = 0; nt < 32; ++nt) {
        const unsigned short* w = (nt < 16) ? wk : wv;
        unsigned short* outp    = (nt < 16) ? kout : vout;
        const int nbase = (nt & 15) * 16;
        v8f acc = vzero();
        #pragma unroll
        for (int ks = 0; ks < 8; ++ks) {
            const int row = lane & 15, koff = (lane < 16) ? 0 : 8;
            FragBF fa;
            fa.q[0] = *(const uint4*)&xt[(wave * 16 + row) * 264 + ks * 32 + koff];
            fa.q[1] = *(const uint4*)&xt[(wave * 16 + row) * 264 + ks * 32 + koff + 16];
            v16bf fb = load_fragB(w + nbase * DD_, DD_, ks * 32);
            acc = wmma_bf16(fa.v, fb, acc);
        }
        const int col = nbase + (lane & 15);
        const size_t r0 = rowbase + wave * 16 + ((lane < 16) ? 0 : 8);
        #pragma unroll
        for (int r = 0; r < 8; ++r)
            outp[(r0 + r) * DD_ + col] = f2bf(acc[r]);
    }
}

// ---------------------------------------------------------------------------
// Kernel 2 (per iter): s_n = LN(slots), slots_bf = bf16(slots),
//                      q = s_n @ Wq^T (padded M=16 per batch), zero sums.
// Grid: 64 blocks (one per batch), 128 threads (4 waves).
// ---------------------------------------------------------------------------
__global__ __launch_bounds__(128) void slots_ln_q_kernel(
    const float* __restrict__ slots, const float* __restrict__ g,
    const float* __restrict__ be, unsigned short* __restrict__ slots_bf,
    const unsigned short* __restrict__ wq, unsigned short* __restrict__ qout,
    float* __restrict__ sums)
{
    __shared__ unsigned short sn[16 * 264];
    const int wave = threadIdx.x >> 5, lane = threadIdx.x & 31;
    const int b = blockIdx.x;

    const int gid = blockIdx.x * 128 + threadIdx.x;
    if (gid < BB_ * NSL) sums[gid] = 0.f;          // reset attn row-sums

    for (int t = 0; t < 2; ++t) {
        const int i = wave * 2 + t;                // slot row 0..7
        const float* xr = slots + (size_t)(b * NSL + i) * DD_;
        float4 a = ((const float4*)xr)[lane * 2];
        float4 c = ((const float4*)xr)[lane * 2 + 1];
        float vals[8] = {a.x, a.y, a.z, a.w, c.x, c.y, c.z, c.w};
        unsigned short raw[8];
        float s = 0.f, s2 = 0.f;
        #pragma unroll
        for (int e = 0; e < 8; ++e) {
            raw[e] = f2bf(vals[e]);
            s += vals[e]; s2 += vals[e] * vals[e];
        }
        *(uint4*)(slots_bf + (size_t)(b * NSL + i) * DD_ + lane * 8) = *(const uint4*)raw;
        s = wave_sum(s); s2 = wave_sum(s2);
        const float mean = s * (1.f / 256.f);
        const float inv  = rsqrtf(s2 * (1.f / 256.f) - mean * mean + 1e-5f);
        unsigned short o[8];
        #pragma unroll
        for (int e = 0; e < 8; ++e)
            o[e] = f2bf((vals[e] - mean) * inv * g[lane * 8 + e] + be[lane * 8 + e]);
        *(uint4*)&sn[i * 264 + lane * 8] = *(const uint4*)o;
    }
    __syncthreads();

    for (int c = 0; c < 4; ++c) {                  // 16 n-tiles over 4 waves
        const int nbase = (wave * 4 + c) * 16;
        v8f acc = vzero();
        #pragma unroll
        for (int ks = 0; ks < 8; ++ks) {
            const int row = lane & 15, koff = (lane < 16) ? 0 : 8;
            FragBF fa;
            fa.q[0] = *(const uint4*)&sn[row * 264 + ks * 32 + koff];
            fa.q[1] = *(const uint4*)&sn[row * 264 + ks * 32 + koff + 16];
            v16bf fb = load_fragB(wq + nbase * DD_, DD_, ks * 32);
            acc = wmma_bf16(fa.v, fb, acc);
        }
        const int col = nbase + (lane & 15);
        const int r0  = (lane < 16) ? 0 : 8;
        #pragma unroll
        for (int r = 0; r < 8; ++r)
            qout[(size_t)(b * 16 + r0 + r) * DD_ + col] = f2bf(acc[r]);
    }
}

// ---------------------------------------------------------------------------
// Kernel 3 (per iter): dots = q @ k^T * scale; softmax over slots; +eps;
//                      write attn (bf16, unnormalized by mean) and row sums.
// Grid: 64*32 blocks, 256 threads (8 waves), each wave one 16-token j-tile.
// After WMMA, lane L (<16) holds dots[i=0..7][j=jbase+L] -> lane-local softmax.
// ---------------------------------------------------------------------------
__global__ __launch_bounds__(256) void dots_softmax_kernel(
    const unsigned short* __restrict__ qb, const unsigned short* __restrict__ kb,
    unsigned short* __restrict__ attn, float* __restrict__ sums)
{
    const int wave = threadIdx.x >> 5, lane = threadIdx.x & 31;
    const int b  = blockIdx.x >> 5;
    const int jg = blockIdx.x & 31;
    const int jbase = jg * 128 + wave * 16;

    const unsigned short* qbase = qb + (size_t)b * 16 * DD_;
    const unsigned short* kp    = kb + ((size_t)b * NN_ + jbase) * DD_;
    __builtin_prefetch(kp + (size_t)(lane & 15) * DD_, 0, 1);

    v8f acc = vzero();
    #pragma unroll
    for (int ks = 0; ks < 8; ++ks) {
        v16bf fa = load_fragA(qbase, DD_, ks * 32);
        v16bf fb = load_fragB(kp,    DD_, ks * 32);
        acc = wmma_bf16(fa, fb, acc);
    }

    float e[8];
    #pragma unroll
    for (int r = 0; r < 8; ++r) e[r] = 0.f;
    if (lane < 16) {
        float m = -1e30f;
        #pragma unroll
        for (int r = 0; r < 8; ++r) { acc[r] *= 0.0625f; m = fmaxf(m, acc[r]); }
        float den = 0.f;
        #pragma unroll
        for (int r = 0; r < 8; ++r) { e[r] = __expf(acc[r] - m); den += e[r]; }
        const float inv = 1.f / den;
        #pragma unroll
        for (int r = 0; r < 8; ++r) {
            e[r] = e[r] * inv + 1e-5f;
            attn[((size_t)b * 16 + r) * NN_ + jbase + lane] = f2bf(e[r]);
        }
    }
    #pragma unroll
    for (int r = 0; r < 8; ++r) {
        const float s = wave_sum(e[r]);
        if (lane == 0) atomicAdd(&sums[b * NSL + r], s);
    }
}

// ---------------------------------------------------------------------------
// Kernel 4 (per iter): updates[b,i,d] = (4096/sums[b,i]) * sum_j attn[i,j]*v[j,d]
// Grid: 64 blocks (per batch), 256 threads (8 waves, 2 d-tiles each).
// v tiles (32 tokens x 256 d) staged through LDS; B frags gathered column-wise.
// ---------------------------------------------------------------------------
__global__ __launch_bounds__(256) void updates_kernel(
    const unsigned short* __restrict__ attn, const unsigned short* __restrict__ vb,
    const float* __restrict__ sums, float* __restrict__ upd,
    unsigned short* __restrict__ upd_bf)
{
    __shared__ unsigned short vt[32 * 264];
    const int wave = threadIdx.x >> 5, lane = threadIdx.x & 31;
    const int b = blockIdx.x;
    const unsigned short* arow = attn + (size_t)b * 16 * NN_;

    v8f acc0 = vzero(), acc1 = vzero();
    for (int ks = 0; ks < 128; ++ks) {
        const int kbv = ks * 32;
        __syncthreads();
        {   // cooperative load: rows kbv..kbv+31, 32B per thread
            const int row = threadIdx.x >> 3;
            const int seg = (threadIdx.x & 7) * 32;
            const uint4* src =
                (const uint4*)(vb + ((size_t)b * NN_ + kbv + row) * DD_ + seg);
            if (ks + 1 < 128)
                __builtin_prefetch(vb + ((size_t)b * NN_ + kbv + 32 + row) * DD_ + seg, 0, 1);
            *(uint4*)&vt[row * 264 + seg]     = src[0];
            *(uint4*)&vt[row * 264 + seg + 8] = src[1];
        }
        __syncthreads();
        v16bf fa = load_fragA(arow, NN_, kbv);
        #pragma unroll
        for (int c = 0; c < 2; ++c) {
            const int nbase = (wave * 2 + c) * 16;
            const int n = nbase + (lane & 15);
            const int joff = (lane < 16) ? 0 : 8;
            FragBF fb;
            #pragma unroll
            for (int t = 0; t < 16; ++t) fb.h[t] = vt[(joff + (t & 7) + (t >> 3) * 16) * 264 + n];
            if (c == 0) acc0 = wmma_bf16(fa, fb.v, acc0);
            else        acc1 = wmma_bf16(fa, fb.v, acc1);
        }
    }
    if (lane < 16) {
        #pragma unroll
        for (int c = 0; c < 2; ++c) {
            const int col = (wave * 2 + c) * 16 + lane;
            #pragma unroll
            for (int r = 0; r < 8; ++r) {
                const float sc = 4096.f / sums[b * NSL + r];
                const float val = ((c == 0) ? acc0[r] : acc1[r]) * sc;
                upd[(size_t)(b * NSL + r) * DD_ + col]    = val;
                upd_bf[(size_t)(b * NSL + r) * DD_ + col] = f2bf(val);
            }
        }
    }
}

// ---------------------------------------------------------------------------
// Kernel 5 (per iter): GRU pre-gates.
//   rz_pre[.,0:512]  = upd@Wih[0:512]^T + slots@Whh[0:512]^T   (fused acc)
//   gin = upd@Wih[512:768]^T ; ghn = slots@Whh[512:768]^T
// Grid: 32 blocks (16-row M tiles over 512 rows), 256 threads (8 waves).
// ---------------------------------------------------------------------------
__global__ __launch_bounds__(256) void gru_gemm_kernel(
    const unsigned short* __restrict__ upd_bf,
    const unsigned short* __restrict__ slots_bf,
    const unsigned short* __restrict__ wih, const unsigned short* __restrict__ whh,
    float* __restrict__ rz_pre, float* __restrict__ gin, float* __restrict__ ghn)
{
    const int wave = threadIdx.x >> 5, lane = threadIdx.x & 31;
    const size_t mbase = (size_t)blockIdx.x * 16;
    const unsigned short* a1 = upd_bf   + mbase * DD_;
    const unsigned short* a2 = slots_bf + mbase * DD_;

    for (int j = wave; j < 48; j += 8) {
        const int nbase = j * 16;
        const size_t r0 = mbase + ((lane < 16) ? 0 : 8);
        const int col = (lane & 15);
        if (j < 32) {
            v8f acc = vzero();
            #pragma unroll
            for (int ks = 0; ks < 8; ++ks) {
                acc = wmma_bf16(load_fragA(a1, DD_, ks * 32),
                                load_fragB(wih + nbase * DD_, DD_, ks * 32), acc);
                acc = wmma_bf16(load_fragA(a2, DD_, ks * 32),
                                load_fragB(whh + nbase * DD_, DD_, ks * 32), acc);
            }
            #pragma unroll
            for (int r = 0; r < 8; ++r)
                rz_pre[(r0 + r) * 512 + nbase + col] = acc[r];
        } else {
            v8f ai = vzero(), ah = vzero();
            #pragma unroll
            for (int ks = 0; ks < 8; ++ks) {
                ai = wmma_bf16(load_fragA(a1, DD_, ks * 32),
                               load_fragB(wih + nbase * DD_, DD_, ks * 32), ai);
                ah = wmma_bf16(load_fragA(a2, DD_, ks * 32),
                               load_fragB(whh + nbase * DD_, DD_, ks * 32), ah);
            }
            const int cc = nbase - 512 + col;
            #pragma unroll
            for (int r = 0; r < 8; ++r) {
                gin[(r0 + r) * DD_ + cc] = ai[r];
                ghn[(r0 + r) * DD_ + cc] = ah[r];
            }
        }
    }
}

// Kernel 6: GRU elementwise combine -> slots_mid
__global__ void gru_combine_kernel(
    const float* __restrict__ rz, const float* __restrict__ gin,
    const float* __restrict__ ghn, const float* __restrict__ bih,
    const float* __restrict__ bhh, const float* __restrict__ hprev,
    float* __restrict__ out)
{
    const int idx = blockIdx.x * 256 + threadIdx.x;
    if (idx >= 512 * DD_) return;
    const int row = idx >> 8, c = idx & 255;
    const float rp = rz[row * 512 + c]       + bih[c]       + bhh[c];
    const float zp = rz[row * 512 + 256 + c] + bih[256 + c] + bhh[256 + c];
    const float r = 1.f / (1.f + __expf(-rp));
    const float z = 1.f / (1.f + __expf(-zp));
    const float n = tanhf(gin[idx] + bih[512 + c] + r * (ghn[idx] + bhh[512 + c]));
    out[idx] = (1.f - z) * n + z * hprev[idx];
}

// Kernel 7: LayerNorm per row (wave per row) -> bf16
__global__ __launch_bounds__(128) void ln_rows_bf16_kernel(
    const float* __restrict__ x, const float* __restrict__ g,
    const float* __restrict__ be, unsigned short* __restrict__ out, int nrows)
{
    const int wave = threadIdx.x >> 5, lane = threadIdx.x & 31;
    const int row = blockIdx.x * 4 + wave;
    if (row >= nrows) return;
    const float* xr = x + (size_t)row * DD_;
    float4 a = ((const float4*)xr)[lane * 2];
    float4 c = ((const float4*)xr)[lane * 2 + 1];
    float vals[8] = {a.x, a.y, a.z, a.w, c.x, c.y, c.z, c.w};
    float s = 0.f, s2 = 0.f;
    #pragma unroll
    for (int e = 0; e < 8; ++e) { s += vals[e]; s2 += vals[e] * vals[e]; }
    s = wave_sum(s); s2 = wave_sum(s2);
    const float mean = s * (1.f / 256.f);
    const float inv  = rsqrtf(s2 * (1.f / 256.f) - mean * mean + 1e-5f);
    unsigned short o[8];
    #pragma unroll
    for (int e = 0; e < 8; ++e)
        o[e] = f2bf((vals[e] - mean) * inv * g[lane * 8 + e] + be[lane * 8 + e]);
    *(uint4*)(out + (size_t)row * DD_ + lane * 8) = *(const uint4*)o;
}

// Kernel 8: y1 = leaky_relu(yln @ W1^T + b1) -> bf16 [512,1024]
__global__ __launch_bounds__(256) void mlp1_kernel(
    const unsigned short* __restrict__ yln, const unsigned short* __restrict__ w1,
    const float* __restrict__ b1, unsigned short* __restrict__ y1)
{
    const int wave = threadIdx.x >> 5, lane = threadIdx.x & 31;
    const size_t mbase = (size_t)blockIdx.x * 16;
    const unsigned short* a = yln + mbase * DD_;
    for (int nt = wave; nt < 64; nt += 8) {
        const int nbase = nt * 16;
        v8f acc = vzero();
        #pragma unroll
        for (int ks = 0; ks < 8; ++ks)
            acc = wmma_bf16(load_fragA(a, DD_, ks * 32),
                            load_fragB(w1 + nbase * DD_, DD_, ks * 32), acc);
        const int col = nbase + (lane & 15);
        const float bias = b1[col];
        const size_t r0 = mbase + ((lane < 16) ? 0 : 8);
        #pragma unroll
        for (int r = 0; r < 8; ++r) {
            float v = acc[r] + bias;
            v = (v > 0.f) ? v : 0.01f * v;
            y1[(r0 + r) * 1024 + col] = f2bf(v);
        }
    }
}

// Kernel 9: out = resid + y1 @ W2^T + b2   (f32 out; last iter -> d_out)
__global__ __launch_bounds__(256) void mlp2_kernel(
    const unsigned short* __restrict__ y1, const unsigned short* __restrict__ w2,
    const float* __restrict__ b2, const float* __restrict__ resid,
    float* __restrict__ out)
{
    const int wave = threadIdx.x >> 5, lane = threadIdx.x & 31;
    const size_t mbase = (size_t)blockIdx.x * 16;
    const unsigned short* a = y1 + mbase * 1024;
    #pragma unroll
    for (int c = 0; c < 2; ++c) {
        const int nbase = (wave * 2 + c) * 16;
        v8f acc = vzero();
        for (int ks = 0; ks < 32; ++ks)
            acc = wmma_bf16(load_fragA(a, 1024, ks * 32),
                            load_fragB(w2 + (size_t)nbase * 1024, 1024, ks * 32), acc);
        const int col = nbase + (lane & 15);
        const float bias = b2[col];
        const size_t r0 = mbase + ((lane < 16) ? 0 : 8);
        #pragma unroll
        for (int r = 0; r < 8; ++r)
            out[(r0 + r) * DD_ + col] = resid[(r0 + r) * DD_ + col] + acc[r] + bias;
    }
}

// ---------------------------------------------------------------------------
// Host launcher
// ---------------------------------------------------------------------------
extern "C" void kernel_launch(void* const* d_in, const int* in_sizes, int n_in,
                              void* d_out, int out_size, void* d_ws, size_t ws_size,
                              hipStream_t stream) {
    (void)in_sizes; (void)n_in; (void)out_size; (void)ws_size;

    const float* inputs   = (const float*)d_in[0];
    const float* slots_in = (const float*)d_in[1];
    const float* Wq  = (const float*)d_in[2];
    const float* Wk  = (const float*)d_in[3];
    const float* Wv  = (const float*)d_in[4];
    const float* Wih = (const float*)d_in[5];
    const float* Whh = (const float*)d_in[6];
    const float* bih = (const float*)d_in[7];
    const float* bhh = (const float*)d_in[8];
    const float* ln_in_g = (const float*)d_in[9];
    const float* ln_in_b = (const float*)d_in[10];
    const float* ln_s_g  = (const float*)d_in[11];
    const float* ln_s_b  = (const float*)d_in[12];
    const float* ln_m_g  = (const float*)d_in[13];
    const float* ln_m_b  = (const float*)d_in[14];
    const float* W1 = (const float*)d_in[15];
    const float* b1 = (const float*)d_in[16];
    const float* W2 = (const float*)d_in[17];
    const float* b2 = (const float*)d_in[18];
    float* out = (float*)d_out;

    // ---- workspace carve-up (bytes) ----
    char* ws = (char*)d_ws;
    size_t off = 0;
    auto carve = [&](size_t bytes) { char* p = ws + off; off += (bytes + 255) & ~(size_t)255; return p; };
    unsigned short* k_bf   = (unsigned short*)carve((size_t)BB_ * NN_ * DD_ * 2);
    unsigned short* v_bf   = (unsigned short*)carve((size_t)BB_ * NN_ * DD_ * 2);
    unsigned short* wq_bf  = (unsigned short*)carve(DD_ * DD_ * 2);
    unsigned short* wk_bf  = (unsigned short*)carve(DD_ * DD_ * 2);
    unsigned short* wv_bf  = (unsigned short*)carve(DD_ * DD_ * 2);
    unsigned short* wih_bf = (unsigned short*)carve(3 * DD_ * DD_ * 2);
    unsigned short* whh_bf = (unsigned short*)carve(3 * DD_ * DD_ * 2);
    unsigned short* w1_bf  = (unsigned short*)carve(4 * DD_ * DD_ * 2);
    unsigned short* w2_bf  = (unsigned short*)carve(4 * DD_ * DD_ * 2);
    float*          slots  = (float*)carve((size_t)BB_ * NSL * DD_ * 4);
    unsigned short* slots_bf = (unsigned short*)carve((size_t)BB_ * NSL * DD_ * 2);
    unsigned short* q_bf   = (unsigned short*)carve((size_t)BB_ * 16 * DD_ * 2);
    unsigned short* attn   = (unsigned short*)carve((size_t)BB_ * 16 * NN_ * 2);
    float*          sums   = (float*)carve(BB_ * NSL * 4);
    float*          upd    = (float*)carve((size_t)BB_ * NSL * DD_ * 4);
    unsigned short* upd_bf = (unsigned short*)carve((size_t)BB_ * NSL * DD_ * 2);
    float*          rz     = (float*)carve((size_t)512 * 512 * 4);
    float*          gin    = (float*)carve((size_t)512 * DD_ * 4);
    float*          ghn    = (float*)carve((size_t)512 * DD_ * 4);
    float*          smid   = (float*)carve((size_t)512 * DD_ * 4);
    unsigned short* yln    = (unsigned short*)carve((size_t)512 * DD_ * 2);
    unsigned short* y1     = (unsigned short*)carve((size_t)512 * 1024 * 2);

    // ---- weight conversion to bf16 ----
    auto cvt = [&](const float* s, unsigned short* d, int n) {
        cvt_bf16_kernel<<<(n + 255) / 256, 256, 0, stream>>>(s, d, n);
    };
    cvt(Wq,  wq_bf,  DD_ * DD_);
    cvt(Wk,  wk_bf,  DD_ * DD_);
    cvt(Wv,  wv_bf,  DD_ * DD_);
    cvt(Wih, wih_bf, 3 * DD_ * DD_);
    cvt(Whh, whh_bf, 3 * DD_ * DD_);
    cvt(W1,  w1_bf,  4 * DD_ * DD_);
    cvt(W2,  w2_bf,  4 * DD_ * DD_);

    bcast_slots_kernel<<<(BB_ * NSL * DD_) / 256, 256, 0, stream>>>(slots_in, slots);

    // ---- fused LN + K/V projection (bf16 K/V caches) ----
    ln_kv_kernel<<<(BB_ * NN_) / 64, 128, 0, stream>>>(
        inputs, ln_in_g, ln_in_b, wk_bf, wv_bf, k_bf, v_bf);

    // ---- 3 slot-attention iterations ----
    for (int it = 0; it < 3; ++it) {
        slots_ln_q_kernel<<<BB_, 128, 0, stream>>>(
            slots, ln_s_g, ln_s_b, slots_bf, wq_bf, q_bf, sums);
        dots_softmax_kernel<<<BB_ * 32, 256, 0, stream>>>(q_bf, k_bf, attn, sums);
        updates_kernel<<<BB_, 256, 0, stream>>>(attn, v_bf, sums, upd, upd_bf);
        gru_gemm_kernel<<<32, 256, 0, stream>>>(
            upd_bf, slots_bf, wih_bf, whh_bf, rz, gin, ghn);
        gru_combine_kernel<<<512, 256, 0, stream>>>(rz, gin, ghn, bih, bhh, slots, smid);
        ln_rows_bf16_kernel<<<128, 128, 0, stream>>>(smid, ln_m_g, ln_m_b, yln, 512);
        mlp1_kernel<<<32, 256, 0, stream>>>(yln, w1_bf, b1, y1);
        mlp2_kernel<<<32, 256, 0, stream>>>(
            y1, w2_bf, b2, smid, (it == 2) ? out : slots);
    }
}
